// LSTM_45526653337950
// MI455X (gfx1250) — matrix operand order repaired and checked by reading, not compiled, not run on hardware
//
#include <hip/hip_runtime.h>
#include <stdint.h>

typedef __attribute__((ext_vector_type(16))) __bf16        v16bf;
typedef __attribute__((ext_vector_type(8)))  float         v8f;
typedef __attribute__((ext_vector_type(4)))  unsigned int  v4u;
typedef __attribute__((ext_vector_type(4)))  float         v4f;
typedef int v4i_vs __attribute__((vector_size(16)));       // matches builtin param type

#define BB 64
#define TT 512
#define II 512
#define HH 512
#define G4 2048          // 4*H
#define MROWS (BB*TT)    // 32768

#define AS1 __attribute__((address_space(1)))
#define AS3 __attribute__((address_space(3)))

#if defined(__HIP_DEVICE_COMPILE__) && __has_builtin(__builtin_amdgcn_global_load_async_to_lds_b128)
#define HAVE_ASYNC 1
#else
#define HAVE_ASYNC 0
#endif

#if HAVE_ASYNC
__device__ __forceinline__ void async_g2l_b128(const void* g, void* l) {
  __builtin_amdgcn_global_load_async_to_lds_b128(
      (AS1 v4i_vs*)(unsigned long long)(uintptr_t)g,   // generic == global addr
      (AS3 v4i_vs*)(unsigned)(uintptr_t)l,             // LDS offset = addr[31:0]
      0, 0);
}
#endif

__device__ __forceinline__ void wait_async0() {
#if defined(__HIP_DEVICE_COMPILE__)
#if __has_builtin(__builtin_amdgcn_s_wait_asynccnt)
  __builtin_amdgcn_s_wait_asynccnt(0);
#else
  asm volatile("s_wait_asynccnt 0" ::: "memory");
#endif
#endif
}

__device__ __forceinline__ unsigned short f2bf(float f) {
  union { float f; unsigned u; } x; x.f = f;
  unsigned u = x.u;
  u += 0x7fffu + ((u >> 16) & 1u);   // round-to-nearest-even
  return (unsigned short)(u >> 16);
}

union FragBF { v16bf v; unsigned short u[16]; v4u q[2]; };

__device__ __forceinline__ void loadB4(const unsigned short* bp, int strideN16,
                                       FragBF& b0, FragBF& b1, FragBF& b2, FragBF& b3) {
  b0.q[0] = *(const v4u*)(bp);                 b0.q[1] = *(const v4u*)(bp + 8);
  b1.q[0] = *(const v4u*)(bp + strideN16);     b1.q[1] = *(const v4u*)(bp + strideN16 + 8);
  b2.q[0] = *(const v4u*)(bp + 2 * strideN16); b2.q[1] = *(const v4u*)(bp + 2 * strideN16 + 8);
  b3.q[0] = *(const v4u*)(bp + 3 * strideN16); b3.q[1] = *(const v4u*)(bp + 3 * strideN16 + 8);
}

__device__ __forceinline__ float sigmf(float x) { return 1.0f / (1.0f + __expf(-x)); }

// ---- convert input_seq f32 -> bf16 (packed pair stores) --------------------
__global__ void k_cvt_x(const float* __restrict__ x, unsigned short* __restrict__ xb, int n2) {
  int i = blockIdx.x * blockDim.x + threadIdx.x;
  if (i >= n2) return;
  unsigned lo = f2bf(x[2 * i]);
  unsigned hi = f2bf(x[2 * i + 1]);
  ((unsigned*)xb)[i] = lo | (hi << 16);
}

// ---- transpose + convert weights: W [512,2048] f32 -> WT [2048,512] bf16 ---
__global__ void k_cvt_wT(const float* __restrict__ w, unsigned short* __restrict__ wt) {
  int i = blockIdx.x * blockDim.x + threadIdx.x;   // over 2048*512
  if (i >= G4 * II) return;
  int n = i / II, k = i % II;
  wt[i] = f2bf(w[(size_t)k * G4 + n]);
}

// ---- Phase 1 GEMM: xproj = Xb @ WxT^T + bx + bh ----------------------------
// Block = 8 waves sharing one 16-row A tile (16KB, contiguous) staged in LDS via
// async-to-LDS; each wave computes a 16x64 output slab (4 accumulators) with a
// register double-buffer on the B fragments (prefetch next K-chunk before MACs).
__global__ void __launch_bounds__(256) k_gemm_xproj(
    const unsigned short* __restrict__ xb,   // [32768,512] bf16
    const unsigned short* __restrict__ wxT,  // [2048,512]  bf16
    const float* __restrict__ bx, const float* __restrict__ bh,
    float* __restrict__ xproj)               // [32768,2048] f32
{
  __shared__ unsigned char sA[16 * II * 2];  // 16KB A tile, row-major [16][512] bf16

  const int mt = blockIdx.x >> 2;                              // same for all 8 waves
  {
    const unsigned char* gsrc = (const unsigned char*)(xb + (size_t)mt * 16 * II);
#if HAVE_ASYNC
    for (int i = threadIdx.x; i < 1024; i += 256)
      async_g2l_b128(gsrc + (size_t)i * 16, sA + (size_t)i * 16);
    wait_async0();
#else
    for (int i = threadIdx.x; i < 1024; i += 256)
      *(v4u*)(sA + (size_t)i * 16) = *(const v4u*)(gsrc + (size_t)i * 16);
#endif
    __syncthreads();
  }

  const int lane = threadIdx.x & 31;
  const int ngrp = (blockIdx.x & 3) * 8 + (threadIdx.x >> 5); // 0..31
  const int nb   = ngrp * 64;
  const int koffA = (lane < 16) ? 0 : 16;  // A frag: K 0..15 | 16..31 contiguous
  const int koffB = (lane < 16) ? 0 : 16;  // B frag: K 0..15 | 16..31 contiguous
  const unsigned short* sa   = (const unsigned short*)sA + (lane & 15) * II + koffA;
  const unsigned short* bcol = wxT + (size_t)(nb + (lane & 15)) * II + koffB;

  v8f a0 = {}, a1 = {}, a2 = {}, a3 = {};
  FragBF cb0, cb1, cb2, cb3;
  loadB4(bcol, 16 * II, cb0, cb1, cb2, cb3);          // preload K-chunk 0
#pragma unroll
  for (int kk = 0; kk < II; kk += 32) {
    FragBF a;
    a.q[0] = *(const v4u*)(sa + kk);                  // ds_load_b128
    a.q[1] = *(const v4u*)(sa + kk + 8);
    FragBF nb0, nb1, nb2, nb3;
    if (kk + 32 < II) {
      loadB4(bcol + kk + 32, 16 * II, nb0, nb1, nb2, nb3);  // prefetch next chunk
    } else {
      nb0 = cb0; nb1 = cb1; nb2 = cb2; nb3 = cb3;
    }
    a0 = __builtin_amdgcn_wmma_f32_16x16x32_bf16(false, a.v, false, cb0.v, (short)0, a0, false, false);
    a1 = __builtin_amdgcn_wmma_f32_16x16x32_bf16(false, a.v, false, cb1.v, (short)0, a1, false, false);
    a2 = __builtin_amdgcn_wmma_f32_16x16x32_bf16(false, a.v, false, cb2.v, (short)0, a2, false, false);
    a3 = __builtin_amdgcn_wmma_f32_16x16x32_bf16(false, a.v, false, cb3.v, (short)0, a3, false, false);
    cb0 = nb0; cb1 = nb1; cb2 = nb2; cb3 = nb3;
  }
  const int col = nb + (lane & 15);
  const int rb  = (lane < 16) ? 0 : 8;
#pragma unroll
  for (int r = 0; r < 8; ++r) {
    size_t o = (size_t)(mt * 16 + rb + r) * G4;
    xproj[o + col     ] = a0[r] + bx[col     ] + bh[col     ];
    xproj[o + col + 16] = a1[r] + bx[col + 16] + bh[col + 16];
    xproj[o + col + 32] = a2[r] + bx[col + 32] + bh[col + 32];
    xproj[o + col + 48] = a3[r] + bx[col + 48] + bh[col + 48];
  }
}

// ---- Phase 2 step GEMM: z = xproj[:,t,:] + h @ Wh --------------------------
// Block = 8 waves sharing one 16-row h tile; block cooperatively converts the
// shared tile f32->bf16 into LDS ONCE, then the K-loop consumes A via
// ds_load_b128 and double-buffers B from L2-resident Wh^T.
__global__ void __launch_bounds__(256) k_step_gemm(
    const float* __restrict__ h,             // [64,512] f32
    const unsigned short* __restrict__ whT,  // [2048,512] bf16
    const float* __restrict__ xproj,         // [32768,2048] f32
    float* __restrict__ z, int t)            // z [64,2048]
{
  __shared__ unsigned int sH[16 * HH / 2];   // 16KB: bf16 pairs, row-major [16][512]

  const int mt = blockIdx.x >> 2;            // 0..3, same for all 8 waves
  {
    const float* hbase = h + (size_t)mt * 16 * HH;
    for (int i = threadIdx.x; i < 16 * HH / 2; i += 256) {
      unsigned lo = f2bf(hbase[2 * i]);
      unsigned hi = f2bf(hbase[2 * i + 1]);
      sH[i] = lo | (hi << 16);               // ds_store_b32
    }
    __syncthreads();
  }

  const int lane = threadIdx.x & 31;
  const int ngrp = (blockIdx.x & 3) * 8 + (threadIdx.x >> 5);
  const int nb   = ngrp * 64;
  const int koffA = (lane < 16) ? 0 : 16;
  const int koffB = (lane < 16) ? 0 : 16;
  const unsigned short* sa   = (const unsigned short*)sH + (lane & 15) * HH + koffA;
  const unsigned short* bcol = whT + (size_t)(nb + (lane & 15)) * HH + koffB;

  v8f a0 = {}, a1 = {}, a2 = {}, a3 = {};
  FragBF cb0, cb1, cb2, cb3;
  loadB4(bcol, 16 * HH, cb0, cb1, cb2, cb3);
#pragma unroll
  for (int kk = 0; kk < HH; kk += 32) {
    FragBF a;
    a.q[0] = *(const v4u*)(sa + kk);          // ds_load_b128
    a.q[1] = *(const v4u*)(sa + kk + 8);
    FragBF nb0, nb1, nb2, nb3;
    if (kk + 32 < HH) {
      loadB4(bcol + kk + 32, 16 * HH, nb0, nb1, nb2, nb3);
    } else {
      nb0 = cb0; nb1 = cb1; nb2 = cb2; nb3 = cb3;
    }
    a0 = __builtin_amdgcn_wmma_f32_16x16x32_bf16(false, a.v, false, cb0.v, (short)0, a0, false, false);
    a1 = __builtin_amdgcn_wmma_f32_16x16x32_bf16(false, a.v, false, cb1.v, (short)0, a1, false, false);
    a2 = __builtin_amdgcn_wmma_f32_16x16x32_bf16(false, a.v, false, cb2.v, (short)0, a2, false, false);
    a3 = __builtin_amdgcn_wmma_f32_16x16x32_bf16(false, a.v, false, cb3.v, (short)0, a3, false, false);
    cb0 = nb0; cb1 = nb1; cb2 = nb2; cb3 = nb3;
  }
  const int col = nb + (lane & 15);
  const int rb  = (lane < 16) ? 0 : 8;
#pragma unroll
  for (int r = 0; r < 8; ++r) {
    int bidx = mt * 16 + rb + r;                       // batch row
    const float* xp = xproj + ((size_t)bidx * TT + t) * G4;
    float* zb = z + (size_t)bidx * G4;
    zb[col     ] = a0[r] + xp[col     ];
    zb[col + 16] = a1[r] + xp[col + 16];
    zb[col + 32] = a2[r] + xp[col + 32];
    zb[col + 48] = a3[r] + xp[col + 48];
  }
}

// ---- Phase 2 elementwise: gates, cell, LayerNorm, h, output ---------------
__global__ void k_step_elem(float* __restrict__ h, float* __restrict__ c,
                            const float* __restrict__ z,
                            const float* __restrict__ gamma, const float* __restrict__ beta,
                            float* __restrict__ out, int t)
{
  __shared__ float s_sum[256], s_sq[256];
  const int b = blockIdx.x, tid = threadIdx.x;
  const float* zb = z + (size_t)b * G4;
  float* cb = c + (size_t)b * HH;
  float* hb = h + (size_t)b * HH;
  float og[2], cn[2];
  float sum = 0.f, sq = 0.f;
#pragma unroll
  for (int j = 0; j < 2; ++j) {
    int hc = tid + j * 256;
    float ig = sigmf(zb[hc]);
    float fg = sigmf(zb[HH + hc]);
    og[j]    = sigmf(zb[2 * HH + hc]);
    float g  = tanhf(zb[3 * HH + hc]);
    float v  = fg * cb[hc] + ig * g;
    cn[j] = v; sum += v; sq += v * v;
  }
  s_sum[tid] = sum; s_sq[tid] = sq;
  __syncthreads();
  for (int off = 128; off > 0; off >>= 1) {
    if (tid < off) { s_sum[tid] += s_sum[tid + off]; s_sq[tid] += s_sq[tid + off]; }
    __syncthreads();
  }
  float mu  = s_sum[0] * (1.0f / HH);
  float var = s_sq[0]  * (1.0f / HH) - mu * mu;
  float inv = rsqrtf(var + 1e-5f);
  float* ob = out + ((size_t)b * TT + t) * HH;
#pragma unroll
  for (int j = 0; j < 2; ++j) {
    int hc = tid + j * 256;
    float cl = (cn[j] - mu) * inv * gamma[hc] + beta[hc];
    cb[hc] = cl;
    float hn = og[j] * tanhf(cl);
    hb[hc] = hn;
    ob[hc] = hn;
  }
}

__global__ void k_zero(float* __restrict__ p, int n) {
  int i = blockIdx.x * blockDim.x + threadIdx.x;
  if (i < n) p[i] = 0.0f;
}

__global__ void k_tail(const float* __restrict__ h, const float* __restrict__ c,
                       float* __restrict__ out) {
  int i = blockIdx.x * blockDim.x + threadIdx.x;
  size_t base = (size_t)BB * TT * HH;
  if (i < BB * HH)            out[base + i] = h[i];
  else if (i < 2 * BB * HH)   out[base + i] = c[i - BB * HH];
}

extern "C" void kernel_launch(void* const* d_in, const int* in_sizes, int n_in,
                              void* d_out, int out_size, void* d_ws, size_t ws_size,
                              hipStream_t stream) {
  (void)in_sizes; (void)n_in; (void)out_size; (void)ws_size;
  const float* x     = (const float*)d_in[0];
  const float* Wx    = (const float*)d_in[1];
  const float* Wh    = (const float*)d_in[2];
  const float* bx    = (const float*)d_in[3];
  const float* bh    = (const float*)d_in[4];
  const float* gamma = (const float*)d_in[5];
  const float* beta  = (const float*)d_in[6];
  float* out = (float*)d_out;

  char* ws = (char*)d_ws;
  size_t off = 0;
  float* xproj          = (float*)(ws + off);           off += (size_t)MROWS * G4 * 4; // 256 MB
  unsigned short* xb    = (unsigned short*)(ws + off);  off += (size_t)MROWS * II * 2; //  32 MB
  unsigned short* wxT   = (unsigned short*)(ws + off);  off += (size_t)G4 * II * 2;    //   2 MB
  unsigned short* whT   = (unsigned short*)(ws + off);  off += (size_t)G4 * II * 2;    //   2 MB
  float* zbuf           = (float*)(ws + off);           off += (size_t)BB * G4 * 4;
  float* hbuf           = (float*)(ws + off);           off += (size_t)BB * HH * 4;
  float* cbuf           = (float*)(ws + off);           off += (size_t)BB * HH * 4;

  // precision conversion + weight transpose (bf16, L2-resident afterwards)
  k_cvt_x <<<(MROWS * II / 2 + 255) / 256, 256, 0, stream>>>(x, xb, MROWS * II / 2);
  k_cvt_wT<<<(G4 * II + 255) / 256, 256, 0, stream>>>(Wx, wxT);
  k_cvt_wT<<<(G4 * II + 255) / 256, 256, 0, stream>>>(Wh, whT);

  // big parallel GEMM: 8192 blocks x 8 waves, A tile async-staged in LDS
  k_gemm_xproj<<<8192, 256, 0, stream>>>(xb, wxT, bx, bh, xproj);

  // init h0 = c0 = 0 (h and c are adjacent in ws)
  k_zero<<<(2 * BB * HH + 255) / 256, 256, 0, stream>>>(hbuf, 2 * BB * HH);

  // sequential recurrence: graph-captured kernel chain (stream order = dependence)
  for (int t = 0; t < TT; ++t) {
    k_step_gemm<<<16, 256, 0, stream>>>(hbuf, whT, xproj, zbuf, t);
    k_step_elem<<<BB, 256, 0, stream>>>(hbuf, cbuf, zbuf, gamma, beta, out, t);
  }

  // final (h_T, c_T) into the output tail
  k_tail<<<(2 * BB * HH + 255) / 256, 256, 0, stream>>>(hbuf, cbuf, out);
}
